// TransformerBlock_28724741275709
// MI455X (gfx1250) — matrix operand order repaired and checked by reading, not compiled, not run on hardware
//
#include <hip/hip_runtime.h>
#include <hip/hip_bf16.h>
#include <math.h>

typedef __bf16 bf16;
typedef __attribute__((ext_vector_type(16))) __bf16 v16bf;
typedef __attribute__((ext_vector_type(8)))  float  v8f;

#define NTOK   4096   // B*S
#define DMODEL 1024
#define SEQ    2048
#define NHEAD  16
#define DHEAD  64
#define DMLP   4096

// Async copy of 16B per lane: cache/memory -> LDS, tracked by ASYNCcnt.
// Per ISA 15.18.3 op 98: the instruction offset is added to both the LDS and the memory address.
__device__ __forceinline__ void async_b128(unsigned lds_addr, const void* gaddr) {
  asm volatile("global_load_async_to_lds_b128 %0, %1, off"
               :: "v"(lds_addr), "v"(gaddr) : "memory");
}
__device__ __forceinline__ void async_b128_pair(unsigned lds_addr, const void* gaddr) {
  asm volatile("global_load_async_to_lds_b128 %0, %1, off\n\t"
               "global_load_async_to_lds_b128 %0, %1, off offset:16"
               :: "v"(lds_addr), "v"(gaddr) : "memory");
}
__device__ __forceinline__ void wait_async0() {
  asm volatile("s_wait_asynccnt 0x0" ::: "memory");
}

// ---------------------------------------------------------------------------
// LayerNorm: one 256-thread block per row of 1024; write bf16 x-hat.
// ---------------------------------------------------------------------------
__global__ __launch_bounds__(256) void ln_kernel(const float* __restrict__ x,
                                                 const float* __restrict__ w,
                                                 const float* __restrict__ b,
                                                 bf16* __restrict__ out) {
  const int row = blockIdx.x;
  const int tid = threadIdx.x;
  const int wv = tid >> 5, ln = tid & 31;
  const float* xr = x + (size_t)row * DMODEL;
  float4 v = *(const float4*)(xr + tid * 4);
  float s = v.x + v.y + v.z + v.w;
#pragma unroll
  for (int off = 16; off >= 1; off >>= 1) { s += __shfl_xor(s, off, 32); }
  __shared__ float r1[8], r2[8];
  if (ln == 0) r1[wv] = s;
  __syncthreads();
  float tot = 0.f;
#pragma unroll
  for (int i = 0; i < 8; ++i) { tot += r1[i]; }
  const float mean = tot * (1.0f / DMODEL);
  const float cx0 = v.x - mean, cx1 = v.y - mean;
  const float cx2 = v.z - mean, cx3 = v.w - mean;
  float sq = cx0 * cx0 + cx1 * cx1 + cx2 * cx2 + cx3 * cx3;
#pragma unroll
  for (int off = 16; off >= 1; off >>= 1) { sq += __shfl_xor(sq, off, 32); }
  if (ln == 0) r2[wv] = sq;
  __syncthreads();
  float tv = 0.f;
#pragma unroll
  for (int i = 0; i < 8; ++i) { tv += r2[i]; }
  const float rstd = rsqrtf(tv * (1.0f / DMODEL) + 1e-5f);
  const int j = tid * 4;
  bf16* o = out + (size_t)row * DMODEL + j;
  o[0] = (bf16)(cx0 * rstd * w[j + 0] + b[j + 0]);
  o[1] = (bf16)(cx1 * rstd * w[j + 1] + b[j + 1]);
  o[2] = (bf16)(cx2 * rstd * w[j + 2] + b[j + 2]);
  o[3] = (bf16)(cx3 * rstd * w[j + 3] + b[j + 3]);
}

// ---------------------------------------------------------------------------
// Pipelined WMMA GEMM: C[M,N] = A[M,K](bf16) * B[K,N](f32->bf16) + bias.
// BMODE 0: B row-major  addr = k*N + n
// BMODE 1: QKV weights (H,D,DH): addr = ((n>>6)*K + k)*64 + (n&63)
// EPI 0: store bf16 | EPI 1: exact GELU then bf16 | EPI 2: f32 addsrc+acc+bias
// Block tile 128x128, K-step 32, ping-pong LDS. A tile: async copy to LDS.
// B tile: register prefetch + f32->bf16 convert after compute. 1 barrier/iter.
// ---------------------------------------------------------------------------
template <int BMODE, int EPI>
__global__ __launch_bounds__(256) void gemm_kernel(
    const bf16* __restrict__ A, const float* __restrict__ Bw,
    const float* __restrict__ bias, const float* addsrc, float* outF,
    bf16* __restrict__ outB, int M, int N, int K) {
  const int m0 = blockIdx.y * 128, n0 = blockIdx.x * 128;
  const int tid = threadIdx.x;
  const int wv = tid >> 5, ln = tid & 31, g = ln >> 4, l16 = ln & 15;
  __shared__ bf16 As[2][128 * 40];   // [m][k], stride 40 halves (pad)
  __shared__ bf16 Bs[2][128 * 40];   // [n][k], stride 40 halves (pad)

  // issue async copies for A tile (128x32 bf16): 2 x b128 per thread
  auto issueA = [&](int k0, int buf) {
#pragma unroll
    for (int it = 0; it < 2; ++it) {
      const int u = tid + it * 256;
      const int r = u >> 2, c8 = (u & 3) * 8;
      const unsigned lds = (unsigned)(size_t)&As[buf][r * 40 + c8];
      async_b128(lds, A + (size_t)(m0 + r) * K + k0 + c8);
    }
  };
  // prefetch B tile (32x128 f32) into registers
  auto loadB = [&](int k0, float4* regs) {
#pragma unroll
    for (int it = 0; it < 4; ++it) {
      const int u = tid + it * 256;
      const int k = u >> 5, n4 = (u & 31) * 4;
      const float* src;
      if (BMODE == 0) {
        src = Bw + (size_t)(k0 + k) * N + n0 + n4;
      } else {
        const int n = n0 + n4;
        src = Bw + ((size_t)(n >> 6) * K + (k0 + k)) * 64 + (n & 63);
      }
      regs[it] = *(const float4*)src;
    }
  };
  // convert + store B tile transposed to [n][k]
  auto storeB = [&](const float4* regs, int buf) {
#pragma unroll
    for (int it = 0; it < 4; ++it) {
      const int u = tid + it * 256;
      const int k = u >> 5, n4 = (u & 31) * 4;
      Bs[buf][(n4 + 0) * 40 + k] = (bf16)regs[it].x;
      Bs[buf][(n4 + 1) * 40 + k] = (bf16)regs[it].y;
      Bs[buf][(n4 + 2) * 40 + k] = (bf16)regs[it].z;
      Bs[buf][(n4 + 3) * 40 + k] = (bf16)regs[it].w;
    }
  };

  const v8f vzero = {};
  v8f acc[8];
#pragma unroll
  for (int i = 0; i < 8; ++i) { acc[i] = vzero; }

  float4 bReg[4];
  // prologue: stage tile 0
  issueA(0, 0);
  loadB(0, bReg);
  storeB(bReg, 0);
  wait_async0();
  __syncthreads();

  const int NK = K / 32;
  for (int i = 0; i < NK; ++i) {
    const int cur = i & 1, nxt = cur ^ 1;
    if (i + 1 < NK) {
      issueA((i + 1) * 32, nxt);     // async: overlaps with WMMA below
      loadB((i + 1) * 32, bReg);     // global loads in flight during WMMA
    }
    // ---- A fragment: lane M=l16; j -> k = (j/8)*16 + g*8 + j%8 ----
    v16bf af;
    bf16* ap = (bf16*)&af;
    const int mr = wv * 16 + l16;
    *(uint4*)(ap)     = *(const uint4*)(&As[cur][mr * 40 + g * 8]);
    *(uint4*)(ap + 8) = *(const uint4*)(&As[cur][mr * 40 + 16 + g * 8]);
#pragma unroll
    for (int nt = 0; nt < 8; ++nt) {
      // B fragment: lane N=nt*16+l16; j -> k = (j/8)*16 + g*8 + j%8
      v16bf bfv;
      bf16* bp = (bf16*)&bfv;
      const int nr = nt * 16 + l16;
      *(uint4*)(bp)     = *(const uint4*)(&Bs[cur][nr * 40 + g * 8]);
      *(uint4*)(bp + 8) = *(const uint4*)(&Bs[cur][nr * 40 + 16 + g * 8]);
      acc[nt] = __builtin_amdgcn_wmma_f32_16x16x32_bf16(
          false, af, false, bfv, (short)0, acc[nt], false, false);
    }
    if (i + 1 < NK) {
      storeB(bReg, nxt);             // other waves still read buffer 'cur'
    }
    wait_async0();
    __syncthreads();
  }
  // ---- epilogue: C/D layout M = r + 8g, N = l16 ----
#pragma unroll
  for (int nt = 0; nt < 8; ++nt) {
    const int n = n0 + nt * 16 + l16;
    const float bv = bias[n];
#pragma unroll
    for (int r = 0; r < 8; ++r) {
      const int m = m0 + wv * 16 + r + 8 * g;
      float val = acc[nt][r] + bv;
      const size_t idx = (size_t)m * N + n;
      if (EPI == 0) {
        outB[idx] = (bf16)val;
      } else if (EPI == 1) {
        val = 0.5f * val * (1.0f + erff(val * 0.70710678118654752f));
        outB[idx] = (bf16)val;
      } else {
        outF[idx] = addsrc[idx] + val;
      }
    }
  }
}

// ---------------------------------------------------------------------------
// Causal flash attention. Grid (S/128, B*H). 8 waves; wave w owns 16 q rows.
// K 64-key tile async-copied to LDS; V staged transposed via registers;
// P bounced via wave-private LDS (C/D layout -> A fragment layout).
// ---------------------------------------------------------------------------
__global__ __launch_bounds__(256) void attn_kernel(const bf16* __restrict__ Qb,
                                                   const bf16* __restrict__ Kb,
                                                   const bf16* __restrict__ Vb,
                                                   bf16* __restrict__ Zb) {
  const int qblk = blockIdx.x;
  const int bh = blockIdx.y;
  const int bb = bh >> 4, h = bh & 15;
  const int tid = threadIdx.x;
  const int wv = tid >> 5, ln = tid & 31, g = ln >> 4, l16 = ln & 15;
  __shared__ bf16 Kt[64 * 72];        // [key][dh]   stride 72
  __shared__ bf16 Vt[64 * 72];        // [dh][key]   stride 72 (transposed)
  __shared__ bf16 Ps[8 * 16 * 72];    // per-wave P staging, stride 72

  const int q0 = qblk * 128 + wv * 16;
  const size_t tok0 = (size_t)bb * SEQ;
  const v8f vzero = {};

  // Q fragments (16 x 64 contract -> two 16x32 A frags) straight from global
  v16bf qf[2];
#pragma unroll
  for (int c = 0; c < 2; ++c) {
    bf16* qp = (bf16*)&qf[c];
    const bf16* src =
        Qb + (tok0 + q0 + l16) * DMODEL + h * 64 + c * 32 + g * 8;
    *(uint4*)(qp)     = *(const uint4*)(src);
    *(uint4*)(qp + 8) = *(const uint4*)(src + 16);
  }
  v8f o[4];
#pragma unroll
  for (int i = 0; i < 4; ++i) { o[i] = vzero; }
  float rmax[8], rsum[8];
#pragma unroll
  for (int r = 0; r < 8; ++r) { rmax[r] = -3.0e38f; rsum[r] = 0.f; }

  bf16* myP = Ps + wv * 16 * 72;
  const int nkb = (qblk + 1) * 2;          // key blocks of 64 up to causal edge
  for (int kb = 0; kb < nkb; ++kb) {
    const int kbase = kb * 64;
    // ---- stage K (async) and transposed V (via regs) ----
    {
      const int key = tid >> 2;
      const int dh0 = (tid & 3) * 16;
      const size_t gbase = (tok0 + kbase + key) * DMODEL + h * 64 + dh0;
      const unsigned ldsK = (unsigned)(size_t)&Kt[key * 72 + dh0];
      async_b128_pair(ldsK, Kb + gbase);   // 32B: offsets 0 and 16
      const bf16* sV = Vb + gbase;
      const uint4 lo = *(const uint4*)(sV);
      const uint4 hi = *(const uint4*)(sV + 8);
      alignas(16) bf16 tmp[16];
      *(uint4*)(tmp)     = lo;
      *(uint4*)(tmp + 8) = hi;
#pragma unroll
      for (int j = 0; j < 16; ++j) { Vt[(dh0 + j) * 72 + key] = tmp[j]; }
      wait_async0();
    }
    __syncthreads();
    // ---- scores S = Q K^T (B[k=dh][n=key] read from Kt) ----
    v8f s[4];
#pragma unroll
    for (int nt = 0; nt < 4; ++nt) {
      s[nt] = vzero;
      if (kbase + nt * 16 <= q0 + 15) {    // wave-uniform: EXEC stays full
#pragma unroll
        for (int dc = 0; dc < 2; ++dc) {
          v16bf kf;
          bf16* kp = (bf16*)&kf;
          const int kr = nt * 16 + l16;
          *(uint4*)(kp)     = *(const uint4*)(&Kt[kr * 72 + dc * 32 + g * 8]);
          *(uint4*)(kp + 8) =
              *(const uint4*)(&Kt[kr * 72 + dc * 32 + 16 + g * 8]);
          s[nt] = __builtin_amdgcn_wmma_f32_16x16x32_bf16(
              false, qf[dc], false, kf, (short)0, s[nt], false, false);
        }
      }
    }
    // ---- scale + causal mask (C layout: q = q0 + r + 8g, key = col) ----
#pragma unroll
    for (int nt = 0; nt < 4; ++nt) {
      const int key = kbase + nt * 16 + l16;
#pragma unroll
      for (int r = 0; r < 8; ++r) {
        const int q = q0 + r + 8 * g;
        const float v = s[nt][r] * 0.125f;
        s[nt][r] = (key > q) ? -3.0e38f : v;
      }
    }
    // ---- online softmax (cross-lane over 16-lane groups) ----
    float mb[8];
#pragma unroll
    for (int r = 0; r < 8; ++r) {
      mb[r] = fmaxf(fmaxf(s[0][r], s[1][r]), fmaxf(s[2][r], s[3][r]));
    }
#pragma unroll
    for (int off = 8; off >= 1; off >>= 1) {
#pragma unroll
      for (int r = 0; r < 8; ++r) {
        mb[r] = fmaxf(mb[r], __shfl_xor(mb[r], off, 32));
      }
    }
#pragma unroll
    for (int r = 0; r < 8; ++r) {
      const float mn = fmaxf(rmax[r], mb[r]);
      const float cf = expf(rmax[r] - mn);
      rsum[r] *= cf;
      rmax[r] = mn;
#pragma unroll
      for (int t = 0; t < 4; ++t) { o[t][r] = o[t][r] * cf; }
    }
    float ls[8];
#pragma unroll
    for (int r = 0; r < 8; ++r) { ls[r] = 0.f; }
#pragma unroll
    for (int nt = 0; nt < 4; ++nt) {
#pragma unroll
      for (int r = 0; r < 8; ++r) {
        const float p = expf(s[nt][r] - rmax[r]);
        s[nt][r] = p;
        ls[r] += p;
      }
    }
#pragma unroll
    for (int off = 8; off >= 1; off >>= 1) {
#pragma unroll
      for (int r = 0; r < 8; ++r) { ls[r] += __shfl_xor(ls[r], off, 32); }
    }
#pragma unroll
    for (int r = 0; r < 8; ++r) { rsum[r] += ls[r]; }
    // ---- P (bf16) to wave-private LDS, reload as A fragments ----
#pragma unroll
    for (int nt = 0; nt < 4; ++nt) {
#pragma unroll
      for (int r = 0; r < 8; ++r) {
        myP[(r + 8 * g) * 72 + nt * 16 + l16] = (bf16)s[nt][r];
      }
    }
    v16bf pf[2];
#pragma unroll
    for (int kk = 0; kk < 2; ++kk) {
      bf16* pp = (bf16*)&pf[kk];
      *(uint4*)(pp)     = *(const uint4*)(myP + l16 * 72 + kk * 32 + g * 8);
      *(uint4*)(pp + 8) =
          *(const uint4*)(myP + l16 * 72 + kk * 32 + 16 + g * 8);
    }
    // ---- O += P * V (B[k=key][n=dh] read from transposed Vt) ----
#pragma unroll
    for (int nt = 0; nt < 4; ++nt) {
#pragma unroll
      for (int kk = 0; kk < 2; ++kk) {
        v16bf vf;
        bf16* vp = (bf16*)&vf;
        const int dr = nt * 16 + l16;
        *(uint4*)(vp)     = *(const uint4*)(&Vt[dr * 72 + kk * 32 + g * 8]);
        *(uint4*)(vp + 8) =
            *(const uint4*)(&Vt[dr * 72 + kk * 32 + 16 + g * 8]);
        o[nt] = __builtin_amdgcn_wmma_f32_16x16x32_bf16(
            false, pf[kk], false, vf, (short)0, o[nt], false, false);
      }
    }
    __syncthreads();
  }
  // ---- finalize: Z[token][h*64+dh] = O / rowsum ----
  float inv[8];
#pragma unroll
  for (int r = 0; r < 8; ++r) { inv[r] = 1.0f / rsum[r]; }
#pragma unroll
  for (int nt = 0; nt < 4; ++nt) {
#pragma unroll
    for (int r = 0; r < 8; ++r) {
      Zb[(tok0 + q0 + r + 8 * g) * DMODEL + h * 64 + nt * 16 + l16] =
          (bf16)(o[nt][r] * inv[r]);
    }
  }
}

// ---------------------------------------------------------------------------
// Orchestration. Workspace layout (requires >= 40 MB):
//   [0,8MB)   x-hat (bf16)  : LN1 output, later reused for LN2 output
//   [8,16MB)  Q (bf16)   [16,24MB) K   [24,32MB) V   [32,40MB) Z
//   [8,40MB)  MLP hidden (bf16) aliases Q/K/V/Z once they are dead
// d_out holds resid_mid after the W_O GEMM, then the final output.
// ---------------------------------------------------------------------------
extern "C" void kernel_launch(void* const* d_in, const int* in_sizes, int n_in,
                              void* d_out, int out_size, void* d_ws,
                              size_t ws_size, hipStream_t stream) {
  const float* resid_pre = (const float*)d_in[0];
  const float* ln1_w = (const float*)d_in[1];
  const float* ln1_b = (const float*)d_in[2];
  const float* W_Q = (const float*)d_in[3];
  const float* b_Q = (const float*)d_in[4];
  const float* W_K = (const float*)d_in[5];
  const float* b_K = (const float*)d_in[6];
  const float* W_V = (const float*)d_in[7];
  const float* b_V = (const float*)d_in[8];
  const float* W_O = (const float*)d_in[9];
  const float* b_O = (const float*)d_in[10];
  const float* ln2_w = (const float*)d_in[11];
  const float* ln2_b = (const float*)d_in[12];
  const float* W_in = (const float*)d_in[13];
  const float* b_in = (const float*)d_in[14];
  const float* W_out = (const float*)d_in[15];
  const float* b_out = (const float*)d_in[16];
  float* out = (float*)d_out;

  char* ws = (char*)d_ws;
  const size_t SZ = (size_t)NTOK * DMODEL * sizeof(bf16);   // 8 MB
  bf16* xhat = (bf16*)(ws);
  bf16* Qb = (bf16*)(ws + SZ);
  bf16* Kb = (bf16*)(ws + 2 * SZ);
  bf16* Vb = (bf16*)(ws + 3 * SZ);
  bf16* Zb = (bf16*)(ws + 4 * SZ);
  bf16* hid = (bf16*)(ws + SZ);   // aliases Q/K/V/Z region (32 MB)

  const dim3 blk(256);
  const dim3 gD(DMODEL / 128, NTOK / 128);   // (8, 32)
  const dim3 gM(DMLP / 128, NTOK / 128);     // (32, 32)
  const dim3 gA(SEQ / 128, 2 * NHEAD);       // (16, 32)

  // 1) LN1
  ln_kernel<<<NTOK, blk, 0, stream>>>(resid_pre, ln1_w, ln1_b, xhat);
  // 2) Q/K/V projections (weights in (H,D,DH) layout -> BMODE 1)
  gemm_kernel<1, 0><<<gD, blk, 0, stream>>>(xhat, W_Q, b_Q, nullptr, nullptr,
                                            Qb, NTOK, DMODEL, DMODEL);
  gemm_kernel<1, 0><<<gD, blk, 0, stream>>>(xhat, W_K, b_K, nullptr, nullptr,
                                            Kb, NTOK, DMODEL, DMODEL);
  gemm_kernel<1, 0><<<gD, blk, 0, stream>>>(xhat, W_V, b_V, nullptr, nullptr,
                                            Vb, NTOK, DMODEL, DMODEL);
  // 3) causal flash attention
  attn_kernel<<<gA, blk, 0, stream>>>(Qb, Kb, Vb, Zb);
  // 4) output projection + residual: d_out = resid_pre + Z*W_O + b_O
  gemm_kernel<0, 2><<<gD, blk, 0, stream>>>(Zb, W_O, b_O, resid_pre, out,
                                            nullptr, NTOK, DMODEL, DMODEL);
  // 5) LN2 on resid_mid (in d_out), reuse xhat buffer
  ln_kernel<<<NTOK, blk, 0, stream>>>(out, ln2_w, ln2_b, xhat);
  // 6) MLP in + exact GELU
  gemm_kernel<0, 1><<<gM, blk, 0, stream>>>(xhat, W_in, b_in, nullptr, nullptr,
                                            hid, NTOK, DMLP, DMODEL);
  // 7) MLP out + residual: d_out += hid*W_out + b_out
  gemm_kernel<0, 2><<<gD, blk, 0, stream>>>(hid, W_out, b_out, out, out,
                                            nullptr, NTOK, DMODEL, DMLP);
  (void)in_sizes; (void)n_in; (void)out_size; (void)ws_size;
}
